// CapsuleFC_30640296689748
// MI455X (gfx1250) — compile-verified, matrix operand order checked by reading, not run on hardware
//
#include <hip/hip_runtime.h>

#define B_ 64
#define N_ 128
#define A_ 64
#define M_ 128
#define D_ 64
#define SCALE_ 0.125f   // 1/sqrt(D)

typedef _Float16 half_t;
typedef _Float16 v8h  __attribute__((ext_vector_type(8)));
typedef _Float16 v16h __attribute__((ext_vector_type(16)));
typedef float    v8f  __attribute__((ext_vector_type(8)));
typedef float    v4f  __attribute__((ext_vector_type(4)));

__device__ __forceinline__ v8f wmma16x16x32(v16h a, v16h b, v8f c) {
  // D = A(16x32 f16) * B(32x16 f16) + C(16x16 f32)
  return __builtin_amdgcn_wmma_f32_16x16x32_f16(false, a, false, b, (short)0, c,
                                                false, false);
}

// Load one 16x32 f16 operand fragment from LDS (row-major, `stride` halves/row).
// lanes 0-15 hold K {0..7}u{16..23} (+32*kstep); lanes 16-31 hold {8..15}u{24..31}.
__device__ __forceinline__ v16h lds_frag(const half_t* base, int row0, int kstep,
                                         int stride, int lane) {
  int r  = row0 + (lane & 15);
  int k0 = kstep * 32 + ((lane >> 4) & 1) * 8;
  const half_t* p = base + r * stride + k0;
  v8h lo = *(const v8h*)(p);
  v8h hi = *(const v8h*)(p + 16);
  return __builtin_shufflevector(lo, hi, 0, 1, 2, 3, 4, 5, 6, 7,
                                 8, 9, 10, 11, 12, 13, 14, 15);
}

// ---------------------------------------------------------------------------
// Kernel 0: ncv = 0 (atomic accumulation target), na = 1
// ---------------------------------------------------------------------------
__global__ __launch_bounds__(256) void caps_init_kernel(float* __restrict__ out) {
  int idx = blockIdx.x * 256 + threadIdx.x;
  const int total = B_ * M_ * D_ + B_ * M_;
  if (idx < total) out[idx] = (idx < B_ * M_ * D_) ? 0.0f : 1.0f;
}

// ---------------------------------------------------------------------------
// Kernel 1: logits[b,n,m] = SCALE * sum_d votes[b,n,m,d]*ncv_prev[b,m,d]
// Each wave owns 2 m's; d-reduction kept in registers across the 4 d-quarters
// and collapsed with shfl_xor at the end (no LDS atomics). Y values are batch-
// prefetched so global latency hides under the WMMAs.
// grid = (N, M/16), block = 256 (8 waves)
// ---------------------------------------------------------------------------
__global__ __launch_bounds__(256)
void caps_logits_kernel(const float* __restrict__ x,        // (B,N,A)
                        const float* __restrict__ w,        // (N,A,M,D)
                        const float* __restrict__ ncv_prev, // (B,M,D)
                        float* __restrict__ logits)         // (B,N,M)
{
  const int n    = blockIdx.x;
  const int mc   = blockIdx.y;          // chunk of 16 m
  const int tid  = threadIdx.x;
  const int lane = tid & 31, wave = tid >> 5;
  const int hi   = (lane >> 4) & 1;

  __shared__ __align__(16) half_t Xh[64 * 72];  // X_n, rows=b, cols=a (padded)

  // stage X_n (64b x 64a) as f16
  {
    int r = tid >> 2;                  // b row 0..63
    int a0 = (tid & 3) * 16;
    const float* xp = x + ((size_t)r * N_ + n) * A_ + a0;
#pragma unroll
    for (int i = 0; i < 4; ++i) {
      v4f v = *(const v4f*)(xp + i * 4);
#pragma unroll
      for (int j = 0; j < 4; ++j) Xh[r * 72 + a0 + i * 4 + j] = (half_t)v[j];
    }
  }
  __syncthreads();

#pragma unroll 1
  for (int mi = 0; mi < 2; ++mi) {
    const int m = mc * 16 + wave * 2 + mi;
    float racc[4][8];
#pragma unroll
    for (int bt = 0; bt < 4; ++bt)
#pragma unroll
      for (int i = 0; i < 8; ++i) racc[bt][i] = 0.0f;

#pragma unroll 1
    for (int dq = 0; dq < 4; ++dq) {
      const int d = dq * 16 + (lane & 15);
      // batch-prefetch Y values for all 4 b-tiles (independent loads)
      float yv[4][8];
      const float* ybase = ncv_prev + (size_t)m * D_ + d;
#pragma unroll
      for (int bt = 0; bt < 4; ++bt)
#pragma unroll
        for (int i = 0; i < 8; ++i)
          yv[bt][i] = ybase[(size_t)(bt * 16 + i + hi * 8) * (M_ * D_)];
      // B fragments straight from global with f32->f16 convert (w read once)
      const float* wbase = w + (((size_t)n * A_) * M_ + m) * (size_t)D_ + d;
      v16h bf[2];
#pragma unroll
      for (int ks = 0; ks < 2; ++ks) {
        int a0 = ks * 32 + hi * 8;
        v16h f;
#pragma unroll
        for (int i = 0; i < 8; ++i) {
          f[i]     = (half_t)wbase[(size_t)(a0 + i)      * (M_ * D_)];
          f[i + 8] = (half_t)wbase[(size_t)(a0 + 16 + i) * (M_ * D_)];
        }
        bf[ks] = f;
      }
#pragma unroll
      for (int bt = 0; bt < 4; ++bt) {
        v16h a0f = lds_frag(Xh, bt * 16, 0, 72, lane);
        v16h a1f = lds_frag(Xh, bt * 16, 1, 72, lane);
        v8f acc = {0, 0, 0, 0, 0, 0, 0, 0};
        acc = wmma16x16x32(a0f, bf[0], acc);
        acc = wmma16x16x32(a1f, bf[1], acc);
#pragma unroll
        for (int i = 0; i < 8; ++i) racc[bt][i] += acc[i] * yv[bt][i];
      }
    }
    // collapse d: sum the 16 lanes of each half (lanes=d, halves=b+8)
#pragma unroll
    for (int bt = 0; bt < 4; ++bt) {
#pragma unroll
      for (int i = 0; i < 8; ++i) {
        float v = racc[bt][i];
        v += __shfl_xor(v, 1, 32);
        v += __shfl_xor(v, 2, 32);
        v += __shfl_xor(v, 4, 32);
        v += __shfl_xor(v, 8, 32);
        racc[bt][i] = v;
      }
      if ((lane & 15) == 0) {
#pragma unroll
        for (int i = 0; i < 8; ++i) {
          int b = bt * 16 + i + hi * 8;
          logits[((size_t)b * N_ + n) * M_ + m] = racc[bt][i] * SCALE_;
        }
      }
    }
  }
}

// ---------------------------------------------------------------------------
// Kernel 2: in-place softmax over m, gate by next_act, renormalize.
// one wave32 per (b,n) row of 128. grid = B*N/8, block = 256.
// ---------------------------------------------------------------------------
__global__ __launch_bounds__(256)
void caps_softmax_kernel(const float* __restrict__ next_act, // (B,M)
                         float* __restrict__ qk)             // (B,N,M)
{
  int lane = threadIdx.x & 31;
  int wave = threadIdx.x >> 5;
  int row  = blockIdx.x * 8 + wave;   // b*N + n
  int b    = row >> 7;
  float* p = qk + (size_t)row * M_;
  float v[4];
#pragma unroll
  for (int j = 0; j < 4; ++j) v[j] = p[j * 32 + lane];
  float mx = fmaxf(fmaxf(v[0], v[1]), fmaxf(v[2], v[3]));
#pragma unroll
  for (int s = 16; s > 0; s >>= 1) mx = fmaxf(mx, __shfl_xor(mx, s, 32));
  float sum = 0.f;
#pragma unroll
  for (int j = 0; j < 4; ++j) { v[j] = __expf(v[j] - mx); sum += v[j]; }
#pragma unroll
  for (int s = 16; s > 0; s >>= 1) sum += __shfl_xor(sum, s, 32);
  float inv = 1.0f / sum;
  float sum2 = 0.f;
#pragma unroll
  for (int j = 0; j < 4; ++j) {
    v[j] = v[j] * inv * next_act[b * M_ + j * 32 + lane];
    sum2 += v[j];
  }
#pragma unroll
  for (int s = 16; s > 0; s >>= 1) sum2 += __shfl_xor(sum2, s, 32);
  float inv2 = 1.0f / (sum2 + 1e-10f);
#pragma unroll
  for (int j = 0; j < 4; ++j) p[j * 32 + lane] = v[j] * inv2;
}

// ---------------------------------------------------------------------------
// Kernel 3: ncv[:,m,:] += Gbig(64 x K) @ Wbig(K x 64), K = 8192 split 4 ways.
// Gbig[b,(n,a)] = qk[b,n,m]*act[b,n]*x[b,n,a]; scale folded at f16 convert.
// Software-pipelined: next n's x/w/scale prefetched into registers while the
// current iteration's WMMAs execute.
// grid = (M, 4), block = 256 (8 waves, 2 persistent C tiles each).
// ---------------------------------------------------------------------------
__global__ __launch_bounds__(256)
void caps_aggregate_kernel(const float* __restrict__ x,    // (B,N,A)
                           const float* __restrict__ cact, // (B,N)
                           const float* __restrict__ w,    // (N,A,M,D)
                           const float* __restrict__ qk,   // (B,N,M)
                           float* __restrict__ ncv)        // (B,M,D) pre-zeroed
{
  const int m   = blockIdx.x;
  const int n0  = blockIdx.y * 32;
  const int tid = threadIdx.x, lane = tid & 31, wave = tid >> 5;
  const int hi  = (lane >> 4) & 1;
  const int bt  = wave >> 1;
  const int dt0 = (wave & 1) * 2;

  __shared__ __align__(16) half_t Xh[64 * 72]; // scaled poses: rows=b, cols=a
  __shared__ __align__(16) half_t Wt[64 * 72]; // w slice transposed: rows=d, cols=a

  v8f acc0 = {0, 0, 0, 0, 0, 0, 0, 0};
  v8f acc1 = {0, 0, 0, 0, 0, 0, 0, 0};

  const int r  = tid >> 2;        // row (b for x, a for w)
  const int c0 = (tid & 3) * 16;  // column start (a for x, d for w)

  v4f xr[4], wr[4];
  float s;
  auto prefetch = [&](int n) {
    s = qk[((size_t)r * N_ + n) * M_ + m] * cact[r * N_ + n];
    const float* xp = x + ((size_t)r * N_ + n) * A_ + c0;
    const float* wp = w + (((size_t)n * A_ + r) * M_ + m) * (size_t)D_ + c0;
#pragma unroll
    for (int i = 0; i < 4; ++i) {
      xr[i] = *(const v4f*)(xp + i * 4);
      wr[i] = *(const v4f*)(wp + i * 4);
    }
  };
  prefetch(n0);

#pragma unroll 1
  for (int ni = 0; ni < 32; ++ni) {
    __syncthreads();               // previous iteration's fragment reads done
#pragma unroll
    for (int i = 0; i < 4; ++i) {
#pragma unroll
      for (int j = 0; j < 4; ++j) {
        Xh[r * 72 + c0 + i * 4 + j] = (half_t)(xr[i][j] * s);
        Wt[(c0 + i * 4 + j) * 72 + r] = (half_t)wr[i][j];  // transpose a<->d
      }
    }
    __syncthreads();
    if (ni + 1 < 32) prefetch(n0 + ni + 1);  // overlaps with WMMAs below
    v16h a0f = lds_frag(Xh, bt * 16, 0, 72, lane);
    v16h a1f = lds_frag(Xh, bt * 16, 1, 72, lane);
    v16h b00 = lds_frag(Wt, dt0 * 16, 0, 72, lane);
    v16h b01 = lds_frag(Wt, dt0 * 16, 1, 72, lane);
    v16h b10 = lds_frag(Wt, (dt0 + 1) * 16, 0, 72, lane);
    v16h b11 = lds_frag(Wt, (dt0 + 1) * 16, 1, 72, lane);
    acc0 = wmma16x16x32(a0f, b00, acc0);
    acc0 = wmma16x16x32(a1f, b01, acc0);
    acc1 = wmma16x16x32(a0f, b10, acc1);
    acc1 = wmma16x16x32(a1f, b11, acc1);
  }
  // combine the 4 n-split partials in global memory
#pragma unroll
  for (int i = 0; i < 8; ++i) {
    int b = bt * 16 + i + hi * 8;
    int d = lane & 15;
    atomicAdd(&ncv[((size_t)b * M_ + m) * D_ + dt0 * 16 + d], acc0[i]);
    atomicAdd(&ncv[((size_t)b * M_ + m) * D_ + (dt0 + 1) * 16 + d], acc1[i]);
  }
}

// ---------------------------------------------------------------------------
extern "C" void kernel_launch(void* const* d_in, const int* in_sizes, int n_in,
                              void* d_out, int out_size, void* d_ws, size_t ws_size,
                              hipStream_t stream) {
  (void)in_sizes; (void)n_in; (void)out_size; (void)d_ws; (void)ws_size;
  const float* x    = (const float*)d_in[0]; // input (B,N,A)
  const float* cact = (const float*)d_in[1]; // current_act (B,N)
  const float* ncvp = (const float*)d_in[2]; // next_capsule_value (B,M,D)
  const float* nact = (const float*)d_in[3]; // next_act (B,M)
  const float* w    = (const float*)d_in[4]; // w (N,A,M,D)
  // d_in[5] = num_iter == 1 (single routing iteration, as in reference)

  float* out = (float*)d_out;
  float* ncv = out;                                // (B,M,D)
  float* qkp = out + B_ * M_ * D_ + B_ * M_;       // (B,N,M); na sits between

  const int total01 = B_ * M_ * D_ + B_ * M_;
  caps_init_kernel<<<dim3((total01 + 255) / 256), dim3(256), 0, stream>>>(out);
  caps_logits_kernel<<<dim3(N_, M_ / 16), dim3(256), 0, stream>>>(x, w, ncvp, qkp);
  caps_softmax_kernel<<<dim3(B_ * N_ / 8), dim3(256), 0, stream>>>(nact, qkp);
  caps_aggregate_kernel<<<dim3(M_, 4), dim3(256), 0, stream>>>(x, cact, w, qkp, ncv);
}
